// KSGraphAttention_5875515261331
// MI455X (gfx1250) — matrix operand order repaired and checked by reading, not compiled, not run on hardware
//
#include <hip/hip_runtime.h>
#include <hip/hip_bf16.h>
#include <cstdint>

typedef __attribute__((ext_vector_type(16))) _Float16 v16h;
typedef __attribute__((ext_vector_type(8)))  float    v8f;

constexpr int B_  = 2;
constexpr int N_  = 4096;
constexpr int D_  = 256;
constexpr int H_  = 4;
constexpr int HD_ = 64;
constexpr int E_  = 65536;
constexpr int MROWS = B_ * N_;                 // 8192
constexpr int MWPR  = N_ / 32;                 // 128 mask words per row
constexpr int MASK_WORDS = N_ * MWPR;          // 524288
constexpr float SCALE   = 0.125f;              // HD^-0.5
constexpr float NEG_BIG = -1e30f;

// ---- WMMA fragment helpers (layouts per CDNA5 ISA 7.12.2, wave32) ----------

// A-matrix 16x32 f16: lane&15 = row M; halves j map to K via packed pattern,
// lanes 16-31 shifted by +8 within each 16-wide K group.
__device__ __forceinline__ int a_k_of(int j, int hi) {
    int vg = j >> 1, p = j & 1;
    int k  = (vg < 4) ? (2 * vg + p) : (16 + 2 * (vg - 4) + p);
    return k + (hi ? 8 : 0);
}

__device__ __forceinline__ v16h load_a_f16(const _Float16* base, int row_stride, int lane) {
    int m = lane & 15, hi = lane >> 4;
    const _Float16* rp = base + (size_t)m * row_stride;
    v16h a;
#pragma unroll
    for (int j = 0; j < 16; ++j) a[j] = rp[a_k_of(j, hi)];
    return a;
}

// B-matrix 32x16 f16: lane&15 = column N; halves j map to K with the same
// pattern as the A-matrix (a_k_of).
__device__ __forceinline__ v16h load_b_f16(const _Float16* base, int k_stride, int lane) {
    int n = lane & 15, hi = lane >> 4;
    const _Float16* cp = base + n;
    v16h b;
#pragma unroll
    for (int j = 0; j < 16; ++j) b[j] = cp[(size_t)a_k_of(j, hi) * k_stride];
    return b;
}

__device__ __forceinline__ v8f wmma16(v16h a, v16h b, v8f c) {
    return __builtin_amdgcn_wmma_f32_16x16x32_f16(false, a, false, b, (short)0, c, false, false);
}

// ---- Kernel 0: convert activations x -> f16 once ----------------------------
__global__ void ks_xcvt(const float* __restrict__ x, _Float16* __restrict__ xh) {
    int i = blockIdx.x * blockDim.x + threadIdx.x;   // 0 .. MROWS*D_
    xh[i] = (_Float16)x[i];
}

// ---- Kernel 1: weight prep — Wt[sel][k][n] = (f16) W[n][k] ------------------
__global__ void ks_wprep(const float* __restrict__ Wq, const float* __restrict__ Wk,
                         const float* __restrict__ Wv, const float* __restrict__ Wo,
                         _Float16* __restrict__ wt) {
    int i = blockIdx.x * blockDim.x + threadIdx.x;       // 0 .. 4*65536
    int sel = i >> 16, r = i & 65535;
    int n = r >> 8, k = r & 255;
    const float* W = (sel == 0) ? Wq : (sel == 1) ? Wk : (sel == 2) ? Wv : Wo;
    wt[(size_t)sel * 65536 + (size_t)k * 256 + n] = (_Float16)W[(size_t)n * 256 + k];
}

// ---- Kernel 2: mask bitmask init (zeros + diagonal bits) --------------------
__global__ void ks_mask_init(unsigned* __restrict__ mask) {
    int idx = blockIdx.x * blockDim.x + threadIdx.x;     // 0 .. 524288
    if (idx >= MASK_WORDS) return;
    int row = idx >> 7, cw = idx & 127;
    unsigned w = ((row >> 5) == cw) ? (1u << (row & 31)) : 0u;
    mask[idx] = w;
}

// ---- Kernel 3: scatter edges: bit (dst, src) --------------------------------
__global__ void ks_mask_edges(const int* __restrict__ ei, unsigned* __restrict__ mask) {
    int i = blockIdx.x * blockDim.x + threadIdx.x;
    if (i >= E_) return;
    int src = ei[i];
    int dst = ei[E_ + i];
    atomicOr(&mask[(size_t)dst * MWPR + (src >> 5)], 1u << (src & 31));
}

// ---- Kernel 4: fused QKV projection, one wave = one 16x16 out tile ----------
__global__ void ks_qkv_proj(const _Float16* __restrict__ xh, const _Float16* __restrict__ wt,
                            const float* __restrict__ bq, const float* __restrict__ bk,
                            const float* __restrict__ bv,
                            _Float16* __restrict__ Qf, _Float16* __restrict__ Ktf,
                            _Float16* __restrict__ Vf) {
    int lane = threadIdx.x;
    int m0 = blockIdx.x * 16;            // global token-row
    int n0 = blockIdx.y * 16;            // output dim
    int sel = blockIdx.z;                // 0=Q 1=K 2=V
    const _Float16* W = wt + (size_t)sel * 65536;
    const float* bias = (sel == 0) ? bq : (sel == 1) ? bk : bv;

    v8f acc = {};
#pragma unroll
    for (int kc = 0; kc < D_; kc += 32) {
        v16h a = load_a_f16(xh + (size_t)m0 * D_ + kc, D_, lane);
        v16h b = load_b_f16(W + (size_t)kc * D_ + n0, D_, lane);
        acc = wmma16(a, b, acc);
    }
    int hi = lane >> 4, nl = lane & 15;
    int col = n0 + nl, h = col >> 6, hd = col & 63;
    float bb = bias[col];
#pragma unroll
    for (int r = 0; r < 8; ++r) {
        int mg = m0 + r + hi * 8;
        int bidx = mg >> 12, tok = mg & (N_ - 1);
        int bh = bidx * H_ + h;
        _Float16 val = (_Float16)(acc[r] + bb);
        if (sel == 0)      Qf[((size_t)bh * N_ + tok) * HD_ + hd] = val;
        else if (sel == 2) Vf[((size_t)bh * N_ + tok) * HD_ + hd] = val;
        else               Ktf[((size_t)bh * HD_ + hd) * N_ + tok] = val;  // pre-transposed
    }
}

// ---- Kernel 5: flash attention, one wave = (b,h, 16-row query tile) ---------
__global__ void ks_flash_attn(const _Float16* __restrict__ Qf, const _Float16* __restrict__ Ktf,
                              const _Float16* __restrict__ Vf, const unsigned* __restrict__ mask,
                              _Float16* __restrict__ Oh) {
    __shared__ _Float16 lds_p[16 * 32];
    int lane = threadIdx.x;
    int q0 = blockIdx.x * 16;
    int bh = blockIdx.y;
    int hi = lane >> 4, nl = lane & 15;
    int kr = (lane & 15) + ((lane >> 4) << 4);

    const _Float16* qb  = Qf  + ((size_t)bh * N_ + q0) * HD_;
    const _Float16* ktb = Ktf + (size_t)bh * HD_ * N_;
    const _Float16* vb  = Vf  + (size_t)bh * N_ * HD_;

    v16h aq0 = load_a_f16(qb, HD_, lane);        // dims 0..31
    v16h aq1 = load_a_f16(qb + 32, HD_, lane);   // dims 32..63

    v8f oacc[4];
#pragma unroll
    for (int nt = 0; nt < 4; ++nt) oacc[nt] = (v8f){};
    float mrun[8], lrun[8];
#pragma unroll
    for (int r = 0; r < 8; ++r) { mrun[r] = NEG_BIG; lrun[r] = 0.f; }

    for (int kt = 0; kt < N_ / 32; ++kt) {
        int k0 = kt * 32;

        // prefetch next key tile (K rows for this lane + V cacheline)
        if (kt + 1 < N_ / 32) {
            int k0n = k0 + 32;
            __builtin_prefetch(ktb + (size_t)kr * N_ + k0n, 0, 0);
            __builtin_prefetch(ktb + (size_t)(kr + 32) * N_ + k0n, 0, 0);
            __builtin_prefetch(vb + (size_t)(k0n + kr) * HD_, 0, 0);
        }

        // S = Q K^T, 16x32 tile as two 16x16 WMMA accumulators
        v8f s0 = {}, s1 = {};
        s0 = wmma16(aq0, load_b_f16(ktb + k0,            N_, lane), s0);
        s0 = wmma16(aq1, load_b_f16(ktb + 32 * N_ + k0,  N_, lane), s0);
        s1 = wmma16(aq0, load_b_f16(ktb + k0 + 16,           N_, lane), s1);
        s1 = wmma16(aq1, load_b_f16(ktb + 32 * N_ + k0 + 16, N_, lane), s1);

        // one mask word covers this row's 32 keys
        unsigned wrow = (lane < 16) ? mask[(size_t)(q0 + lane) * MWPR + kt] : 0u;

        float p0[8], p1[8];
#pragma unroll
        for (int r = 0; r < 8; ++r) {
            unsigned w = __shfl(wrow, r + hi * 8, 32);
            float v0 = s0[r] * SCALE, v1 = s1[r] * SCALE;
            if (!((w >> nl) & 1u))        v0 = NEG_BIG;
            if (!((w >> (16 + nl)) & 1u)) v1 = NEG_BIG;
            float mx = fmaxf(v0, v1);
#pragma unroll
            for (int off = 8; off >= 1; off >>= 1) mx = fmaxf(mx, __shfl_xor(mx, off, 16));
            float mnew  = fmaxf(mrun[r], mx);
            float alpha = __expf(mrun[r] - mnew);
            float e0 = (v0 <= 0.5f * NEG_BIG) ? 0.f : __expf(v0 - mnew);
            float e1 = (v1 <= 0.5f * NEG_BIG) ? 0.f : __expf(v1 - mnew);
            float rs = e0 + e1;
#pragma unroll
            for (int off = 8; off >= 1; off >>= 1) rs += __shfl_xor(rs, off, 16);
            lrun[r] = lrun[r] * alpha + rs;
            mrun[r] = mnew;
            p0[r] = e0; p1[r] = e1;
#pragma unroll
            for (int nt = 0; nt < 4; ++nt) oacc[nt][r] *= alpha;
        }

        // transpose P (C layout) -> A layout through LDS
        __syncthreads();
#pragma unroll
        for (int r = 0; r < 8; ++r) {
            int m = r + hi * 8;
            lds_p[m * 32 + nl]      = (_Float16)p0[r];
            lds_p[m * 32 + 16 + nl] = (_Float16)p1[r];
        }
        __syncthreads();
        v16h ap = load_a_f16(lds_p, 32, lane);

        // O += P V  (K-dim = 32 keys, 4 column tiles of HD=64)
#pragma unroll
        for (int nt = 0; nt < 4; ++nt)
            oacc[nt] = wmma16(ap, load_b_f16(vb + (size_t)k0 * HD_ + nt * 16, HD_, lane), oacc[nt]);
    }

    int bidx = bh >> 2, h = bh & 3;
#pragma unroll
    for (int nt = 0; nt < 4; ++nt)
#pragma unroll
        for (int r = 0; r < 8; ++r) {
            int tok = q0 + r + hi * 8;
            float l = lrun[r];
            float val = (l > 0.f) ? oacc[nt][r] / l : 0.f;
            Oh[((size_t)bidx * N_ + tok) * D_ + h * HD_ + nt * 16 + nl] = (_Float16)val;
        }
}

// ---- Kernel 6: output projection + bias + residual --------------------------
__global__ void ks_out_proj(const _Float16* __restrict__ Oh, const _Float16* __restrict__ wto,
                            const float* __restrict__ bo, const float* __restrict__ x,
                            float* __restrict__ tmp) {
    int lane = threadIdx.x;
    int m0 = blockIdx.x * 16, n0 = blockIdx.y * 16;
    v8f acc = {};
#pragma unroll
    for (int kc = 0; kc < D_; kc += 32) {
        v16h a = load_a_f16(Oh + (size_t)m0 * D_ + kc, D_, lane);
        v16h b = load_b_f16(wto + (size_t)kc * D_ + n0, D_, lane);
        acc = wmma16(a, b, acc);
    }
    int hi = lane >> 4, nl = lane & 15, col = n0 + nl;
    float bb = bo[col];
#pragma unroll
    for (int r = 0; r < 8; ++r) {
        int mg = m0 + r + hi * 8;
        tmp[(size_t)mg * D_ + col] = acc[r] + bb + x[(size_t)mg * D_ + col];
    }
}

// ---- Kernel 7: row LayerNorm (one wave per row) -----------------------------
__global__ void ks_layernorm(const float* __restrict__ tmp, const float* __restrict__ gamma,
                             const float* __restrict__ beta, float* __restrict__ out) {
    int row = blockIdx.x, lane = threadIdx.x;
    const float* rp = tmp + (size_t)row * D_;
    float v[8], s = 0.f;
#pragma unroll
    for (int i = 0; i < 8; ++i) { v[i] = rp[lane + i * 32]; s += v[i]; }
#pragma unroll
    for (int off = 16; off >= 1; off >>= 1) s += __shfl_xor(s, off, 32);
    float mu = s * (1.f / D_);
    float q = 0.f;
#pragma unroll
    for (int i = 0; i < 8; ++i) { float d = v[i] - mu; q += d * d; }
#pragma unroll
    for (int off = 16; off >= 1; off >>= 1) q += __shfl_xor(q, off, 32);
    float rstd = rsqrtf(q * (1.f / D_) + 1e-5f);
#pragma unroll
    for (int i = 0; i < 8; ++i) {
        int c = lane + i * 32;
        out[(size_t)row * D_ + c] = (v[i] - mu) * rstd * gamma[c] + beta[c];
    }
}

// ---- launch -----------------------------------------------------------------
extern "C" void kernel_launch(void* const* d_in, const int* in_sizes, int n_in,
                              void* d_out, int out_size, void* d_ws, size_t ws_size,
                              hipStream_t stream) {
    (void)in_sizes; (void)n_in; (void)out_size; (void)ws_size;
    const float* x   = (const float*)d_in[0];
    const int*   ei  = (const int*)d_in[1];
    const float* Wq  = (const float*)d_in[2];
    const float* bq  = (const float*)d_in[3];
    const float* Wk  = (const float*)d_in[4];
    const float* bk  = (const float*)d_in[5];
    const float* Wv  = (const float*)d_in[6];
    const float* bv  = (const float*)d_in[7];
    const float* Wo  = (const float*)d_in[8];
    const float* bo  = (const float*)d_in[9];
    const float* gamma = (const float*)d_in[10];
    const float* beta  = (const float*)d_in[11];
    float* out = (float*)d_out;

    char* ws = (char*)d_ws;
    size_t off = 0;
    _Float16* wt16 = (_Float16*)(ws + off); off += (size_t)4 * 65536 * sizeof(_Float16); // 512 KB
    unsigned* mask = (unsigned*)(ws + off); off += (size_t)MASK_WORDS * 4;               // 2 MB
    _Float16* xh   = (_Float16*)(ws + off); off += (size_t)MROWS * D_ * 2;               // 4 MB
    _Float16* Qf   = (_Float16*)(ws + off); off += (size_t)B_ * H_ * N_ * HD_ * 2;       // 4 MB
    _Float16* Ktf  = (_Float16*)(ws + off); off += (size_t)B_ * H_ * N_ * HD_ * 2;       // 4 MB
    _Float16* Vf   = (_Float16*)(ws + off); off += (size_t)B_ * H_ * N_ * HD_ * 2;       // 4 MB
    _Float16* Oh   = (_Float16*)(ws + off); off += (size_t)MROWS * D_ * 2;               // 4 MB
    float*    tmp  = (float*)(ws + off);    off += (size_t)MROWS * D_ * 4;               // 8 MB

    ks_xcvt<<<(MROWS * D_) / 256, 256, 0, stream>>>(x, xh);
    ks_wprep<<<(4 * 65536) / 256, 256, 0, stream>>>(Wq, Wk, Wv, Wo, wt16);
    ks_mask_init<<<MASK_WORDS / 256, 256, 0, stream>>>(mask);
    ks_mask_edges<<<E_ / 256, 256, 0, stream>>>(ei, mask);
    ks_qkv_proj<<<dim3(MROWS / 16, D_ / 16, 3), 32, 0, stream>>>(xh, wt16, bq, bk, bv, Qf, Ktf, Vf);
    ks_flash_attn<<<dim3(N_ / 16, B_ * H_), 32, 0, stream>>>(Qf, Ktf, Vf, mask, Oh);
    ks_out_proj<<<dim3(MROWS / 16, D_ / 16), 32, 0, stream>>>(Oh, wt16 + (size_t)3 * 65536, bo, x, tmp);
    ks_layernorm<<<MROWS, 32, 0, stream>>>(tmp, gamma, beta, out);
}
